// HOT_62466004353037
// MI455X (gfx1250) — matrix operand order repaired and checked by reading, not compiled, not run on hardware
//
#include <hip/hip_runtime.h>
#include <hip/hip_bf16.h>
#include <math.h>

typedef _Float16 v8h  __attribute__((ext_vector_type(8)));
typedef _Float16 v16h __attribute__((ext_vector_type(16)));
typedef float    v8f  __attribute__((ext_vector_type(8)));

constexpr int DDIM  = 1024;
constexpr int BI    = 128;
constexpr int BT    = 128;
constexpr int RTOK  = 36;
constexpr int WTOK  = 50;
constexpr int R_PAD = 48;   // 37 valid rows padded to 3 WMMA tiles
constexpr int W_PAD = 64;   // 51 valid cols padded to 4 WMMA tiles
constexpr int R1    = 37;
constexpr int W1    = 51;
constexpr int LDSW  = 65;   // LDS row stride (floats), avoids 64-bank conflicts
#define EPSF  1e-6f
#define LAMBF 0.1f

// ---------------------------------------------------------------------------
// Kernel 1: prepend CLS, add EPS to tokens, L2-normalize along D, emit f16,
// zero-fill padding rows so WMMA tiles need no masking.
// One 256-thread block per output row (4 elements per thread).
// ---------------------------------------------------------------------------
__global__ __launch_bounds__(256) void hot_normalize_rows(
    const float* __restrict__ cls, const float* __restrict__ toks,
    int ntok, int rows_pad, _Float16* __restrict__ dst)
{
    __shared__ float red[256];
    const int row = blockIdx.x;
    const int b   = row / rows_pad;
    const int r   = row % rows_pad;
    const int tid = threadIdx.x;
    _Float16* drow = dst + (size_t)row * DDIM;

    if (r > ntok) {                      // zero padding row (uniform branch)
        #pragma unroll
        for (int j = 0; j < 4; ++j) drow[tid + 256 * j] = (_Float16)0.f;
        return;
    }
    const float* src;
    float add;
    if (r == 0) { src = cls + (size_t)b * DDIM;                       add = 0.f;  }
    else        { src = toks + ((size_t)b * ntok + (r - 1)) * DDIM;   add = EPSF; }

    float x[4];
    float ss = 0.f;
    #pragma unroll
    for (int j = 0; j < 4; ++j) {
        x[j] = src[tid + 256 * j] + add;
        ss += x[j] * x[j];
    }
    red[tid] = ss;
    __syncthreads();
    for (int s = 128; s > 0; s >>= 1) {
        if (tid < s) red[tid] += red[tid + s];
        __syncthreads();
    }
    const float inv = rsqrtf(red[0]);
    #pragma unroll
    for (int j = 0; j < 4; ++j) drow[tid + 256 * j] = (_Float16)(x[j] * inv);
}

// ---------------------------------------------------------------------------
// Kernel 2: one block per (img i, cap t). 4 wave32 waves compute the padded
// 48x64 sim tile via v_wmma_f32_16x16x32_f16 (wave w = col-tile w, 3 row-tile
// accumulators so the B operand is reused 3x per K-step). Sims land in LDS,
// then masked Sinkhorn (3 iters) + weighted reduction produce out[i,t].
// ---------------------------------------------------------------------------
__global__ __launch_bounds__(128) void hot_sinkhorn_sims(
    const _Float16* __restrict__ An, const _Float16* __restrict__ Bn,
    const int* __restrict__ img_lens, const int* __restrict__ cap_lens,
    float* __restrict__ out)
{
    __shared__ float S[R_PAD * LDSW];   // fg_sims tile
    __shared__ float P[R_PAD * LDSW];   // transport plan
    __shared__ float rs[R_PAD];
    __shared__ float cs[W_PAD];
    __shared__ float red[128];

    const int t    = blockIdx.x;
    const int i    = blockIdx.y;
    const int tid  = threadIdx.x;
    const int lane = tid & 31;
    const int wave = tid >> 5;
    const int hl   = lane >> 4;   // half-wave id
    const int l15  = lane & 15;

    // A (16-bit 16x32): lane holds row l15; K = {k0..k0+7} U {k0+16..k0+23}, k0 = 8*hl
    const _Float16* a0 = An + ((size_t)i * R_PAD + l15) * DDIM + hl * 8;
    const _Float16* a1 = a0 + (size_t)16 * DDIM;
    const _Float16* a2 = a0 + (size_t)32 * DDIM;
    // B (16-bit 32x16): lane holds col l15 (a caption row); K = {k0..k0+7} U {k0+16..k0+23}, k0 = 8*hl
    const _Float16* bp = Bn + ((size_t)t * W_PAD + wave * 16 + l15) * DDIM + hl * 8;

    v8f c0 = {}, c1 = {}, c2 = {};
#pragma unroll 4
    for (int kk = 0; kk < DDIM; kk += 32) {
        v8h bl = *(const v8h*)(bp + kk);
        v8h bh = *(const v8h*)(bp + kk + 16);
        v16h b = __builtin_shufflevector(bl, bh, 0,1,2,3,4,5,6,7,8,9,10,11,12,13,14,15);
        v8h l0 = *(const v8h*)(a0 + kk);
        v8h h0 = *(const v8h*)(a0 + kk + 16);
        v16h A0 = __builtin_shufflevector(l0, h0, 0,1,2,3,4,5,6,7,8,9,10,11,12,13,14,15);
        v8h l1 = *(const v8h*)(a1 + kk);
        v8h h1 = *(const v8h*)(a1 + kk + 16);
        v16h A1 = __builtin_shufflevector(l1, h1, 0,1,2,3,4,5,6,7,8,9,10,11,12,13,14,15);
        v8h l2 = *(const v8h*)(a2 + kk);
        v8h h2 = *(const v8h*)(a2 + kk + 16);
        v16h A2 = __builtin_shufflevector(l2, h2, 0,1,2,3,4,5,6,7,8,9,10,11,12,13,14,15);
        c0 = __builtin_amdgcn_wmma_f32_16x16x32_f16(false, A0, false, b, (short)0, c0, false, false);
        c1 = __builtin_amdgcn_wmma_f32_16x16x32_f16(false, A1, false, b, (short)0, c1, false, false);
        c2 = __builtin_amdgcn_wmma_f32_16x16x32_f16(false, A2, false, b, (short)0, c2, false, false);
    }

    // C layout: element v of lane l sits at (row 8*hl + v, col l15) of the tile
    {
        const int col = wave * 16 + l15;
        #pragma unroll
        for (int v = 0; v < 8; ++v) {
            const int rr = 8 * hl + v;
            S[(rr     ) * LDSW + col] = c0[v];
            S[(rr + 16) * LDSW + col] = c1[v];
            S[(rr + 32) * LDSW + col] = c2[v];
        }
    }
    __syncthreads();

    const int il = img_lens[i] + 1;       // <= 37
    const int cl = cap_lens[t] + 1;       // <= 51
    const float rmarg = 1.f / (float)il;
    const float cmarg = 1.f / (float)cl;

    // ---- P = mask * exp(-(1-S)/lamb); normalize by global sum
    float psum = 0.f;
    for (int idx = tid; idx < R1 * W1; idx += 128) {
        const int r = idx / W1, w = idx - r * W1;
        float p = 0.f;
        if (r < il && w < cl) p = __expf((S[r * LDSW + w] - 1.f) * (1.f / LAMBF));
        P[r * LDSW + w] = p;
        psum += p;
    }
    red[tid] = psum;
    __syncthreads();
    for (int s = 64; s > 0; s >>= 1) {
        if (tid < s) red[tid] += red[tid + s];
        __syncthreads();
    }
    const float inv_tot = 1.f / (red[0] + EPSF);
    for (int idx = tid; idx < R1 * W1; idx += 128) {
        const int r = idx / W1, w = idx - r * W1;
        P[r * LDSW + w] *= inv_tot;
    }
    __syncthreads();

    // ---- 3 Bregman iterations: row rescale to r-marginal, col rescale to c-marginal
    for (int it = 0; it < 3; ++it) {
        if (tid < R1) {
            float u = EPSF;
            for (int w = 0; w < W1; ++w) u += P[tid * LDSW + w];
            rs[tid] = ((tid < il) ? rmarg : 0.f) / u;
        }
        __syncthreads();
        for (int idx = tid; idx < R1 * W1; idx += 128) {
            const int r = idx / W1, w = idx - r * W1;
            P[r * LDSW + w] *= rs[r];
        }
        __syncthreads();
        if (tid < W1) {
            float v = EPSF;
            for (int r = 0; r < R1; ++r) v += P[r * LDSW + tid];
            cs[tid] = ((tid < cl) ? cmarg : 0.f) / v;
        }
        __syncthreads();
        for (int idx = tid; idx < R1 * W1; idx += 128) {
            const int r = idx / W1, w = idx - r * W1;
            P[r * LDSW + w] *= cs[w];
        }
        __syncthreads();
    }

    // ---- sims[i,t] = sum(S * P) over masked region (masked P entries are 0)
    float acc = 0.f;
    for (int idx = tid; idx < R1 * W1; idx += 128) {
        const int r = idx / W1, w = idx - r * W1;
        acc += S[r * LDSW + w] * P[r * LDSW + w];
    }
    red[tid] = acc;
    __syncthreads();
    for (int s = 64; s > 0; s >>= 1) {
        if (tid < s) red[tid] += red[tid + s];
        __syncthreads();
    }
    if (tid == 0) out[(size_t)i * BT + t] = red[0];
}

// ---------------------------------------------------------------------------
extern "C" void kernel_launch(void* const* d_in, const int* in_sizes, int n_in,
                              void* d_out, int out_size, void* d_ws, size_t ws_size,
                              hipStream_t stream)
{
    (void)in_sizes; (void)n_in; (void)out_size; (void)ws_size;
    const float* img_cls  = (const float*)d_in[0];
    const float* imgs     = (const float*)d_in[1];
    const float* cap_cls  = (const float*)d_in[2];
    const float* caps     = (const float*)d_in[3];
    const int*   img_lens = (const int*)d_in[4];
    const int*   cap_lens = (const int*)d_in[5];
    float* out = (float*)d_out;

    _Float16* An = (_Float16*)d_ws;                       // [128*48, 1024] f16
    _Float16* Bn = An + (size_t)BI * R_PAD * DDIM;        // [128*64, 1024] f16

    hot_normalize_rows<<<BI * R_PAD, 256, 0, stream>>>(img_cls, imgs, RTOK, R_PAD, An);
    hot_normalize_rows<<<BT * W_PAD, 256, 0, stream>>>(cap_cls, caps, WTOK, W_PAD, Bn);
    hot_sinkhorn_sims<<<dim3(BT, BI), 128, 0, stream>>>(An, Bn, img_lens, cap_lens, out);
}
